// HeteroGNN_52561809768706
// MI455X (gfx1250) — compile-verified, matrix operand order checked by reading, not run on hardware
//
#include <hip/hip_runtime.h>
#include <hip/hip_bf16.h>

// ---------------- problem dims ----------------
#define N_AUTHOR 50000
#define N_PAPER  25000
#define E_COLLAB 800000
#define E_WRITES 800000
#define D_AUTHOR 128
#define HID      128
#define OUT_DIM  64
#define TOK      350
#define D_BERT   768
#define FEAT     12

#define MA_PAD   50048   // 50000 -> multiple of 64
#define MP_PAD   25024   // 25000 -> multiple of 64
#define K_SAGE0  928     // 128 (mean) + 768 (pooler) + 12 (feat) + 20 pad  (mult of 32)
#define K_SAGE1  256     // 128 (amean) + 128 (p1)

typedef __bf16 bf16;
typedef __attribute__((ext_vector_type(16))) __bf16 v16bf;
typedef __attribute__((ext_vector_type(8)))  float  v8f;
typedef __attribute__((__vector_size__(4 * sizeof(int)))) int v4i;

// ---- CDNA5 async global->LDS feature detection ----
#if defined(__has_builtin)
#  if __has_builtin(__builtin_amdgcn_global_load_async_to_lds_b128)
#    define HAVE_ASYNC_COPY 1
#  endif
#  if __has_builtin(__builtin_amdgcn_s_wait_asynccnt)
#    define HAVE_ASYNC_WAIT 1
#  endif
#endif
#ifndef HAVE_ASYNC_COPY
#  define HAVE_ASYNC_COPY 0
#endif
#ifndef HAVE_ASYNC_WAIT
#  define HAVE_ASYNC_WAIT 0
#endif

// 16-byte global -> LDS copy. Async DMA path on CDNA5 (ASYNCcnt), sync fallback.
__device__ __forceinline__ void copy_b128_g2l(const bf16* g, bf16* l) {
#if HAVE_ASYNC_COPY
  __builtin_amdgcn_global_load_async_to_lds_b128((v4i*)g, (v4i*)l, 0, 0);
#else
  *(uint4*)l = *(const uint4*)g;
#endif
}

__device__ __forceinline__ void wait_async_le2() {
#if HAVE_ASYNC_COPY
#  if HAVE_ASYNC_WAIT
  __builtin_amdgcn_s_wait_asynccnt(2);
#  else
  asm volatile("s_wait_asynccnt 0x2" ::: "memory");
#  endif
#endif
}
__device__ __forceinline__ void wait_async_le0() {
#if HAVE_ASYNC_COPY
#  if HAVE_ASYNC_WAIT
  __builtin_amdgcn_s_wait_asynccnt(0);
#  else
  asm volatile("s_wait_asynccnt 0x0" ::: "memory");
#  endif
#endif
}

static inline int ceil_div(long a, long b) { return (int)((a + b - 1) / b); }

// ---------------- elementwise / utility kernels ----------------
__global__ void zero_f32_kernel(float* __restrict__ p, long n) {
  long i = (long)blockIdx.x * blockDim.x + threadIdx.x;
  if (i < n) p[i] = 0.0f;
}
__global__ void zero_bf16_kernel(bf16* __restrict__ p, long n) {
  long i = (long)blockIdx.x * blockDim.x + threadIdx.x;
  if (i < n) p[i] = (bf16)0.0f;
}

__global__ void count_dst_kernel(const int* __restrict__ dst, float* __restrict__ acc, int E) {
  int e = blockIdx.x * blockDim.x + threadIdx.x;
  if (e < E) atomicAdd(&acc[dst[e]], 1.0f);
}

__global__ void rsqrt_kernel(float* __restrict__ d, int n) {
  int i = blockIdx.x * blockDim.x + threadIdx.x;
  if (i < n) d[i] = rsqrtf(d[i] + 1.0f);
}

__global__ void cvt_rows_kernel(const float* __restrict__ src, bf16* __restrict__ dst,
                                int M, int Mpad, int C) {
  long gid = (long)blockIdx.x * blockDim.x + threadIdx.x;
  if (gid >= (long)Mpad * C) return;
  int row = (int)(gid / C);
  int c   = (int)(gid % C);
  float v = (row < M) ? src[(long)row * C + c] : 0.0f;
  dst[gid] = (bf16)v;
}

__global__ void pack_bt_kernel(const float* __restrict__ W, bf16* __restrict__ Bt,
                               int ldb, int koff, int K, int N) {
  long gid = (long)blockIdx.x * blockDim.x + threadIdx.x;
  if (gid >= (long)K * N) return;
  int k = (int)(gid / N);
  int n = (int)(gid % N);
  Bt[(long)n * ldb + koff + k] = (bf16)W[gid];
}

__global__ void scatter_gcn_kernel(const int* __restrict__ src, const int* __restrict__ dst,
                                   const float* __restrict__ xw, const float* __restrict__ dinv,
                                   float* __restrict__ agg, int E) {
  long gid = (long)blockIdx.x * blockDim.x + threadIdx.x;
  if (gid >= (long)E * 32) return;
  int e = (int)(gid >> 5);
  int c = (int)(gid & 31) << 2;
  int s = src[e], d = dst[e];
  float coef = dinv[s] * dinv[d];
  const float4 v = *(const float4*)(xw + (long)s * 128 + c);
  float* o = agg + (long)d * 128 + c;
  atomicAdd(o + 0, v.x * coef);
  atomicAdd(o + 1, v.y * coef);
  atomicAdd(o + 2, v.z * coef);
  atomicAdd(o + 3, v.w * coef);
}

__global__ void scatter_sum_kernel(const int* __restrict__ src, const int* __restrict__ dst,
                                   const float* __restrict__ x, float* __restrict__ psum, int E) {
  long gid = (long)blockIdx.x * blockDim.x + threadIdx.x;
  if (gid >= (long)E * 32) return;
  int e = (int)(gid >> 5);
  int c = (int)(gid & 31) << 2;
  int s = src[e], d = dst[e];
  const float4 v = *(const float4*)(x + (long)s * 128 + c);
  float* o = psum + (long)d * 128 + c;
  atomicAdd(o + 0, v.x);
  atomicAdd(o + 1, v.y);
  atomicAdd(o + 2, v.z);
  atomicAdd(o + 3, v.w);
}

__global__ void gcn_finalize_kernel(float* __restrict__ agg, const float* __restrict__ xw,
                                    const float* __restrict__ dinv, const float* __restrict__ b,
                                    int M) {
  long gid = (long)blockIdx.x * blockDim.x + threadIdx.x;
  if (gid >= (long)M * 128) return;
  int row = (int)(gid >> 7);
  int c   = (int)(gid & 127);
  float di = dinv[row];
  float v = agg[gid] + xw[gid] * di * di + b[c];
  agg[gid] = v > 0.0f ? v : 0.0f;
}

__global__ void mean_cvt_kernel(const float* __restrict__ psum, const float* __restrict__ cnt,
                                bf16* __restrict__ out, int ldo, int M, int Mpad) {
  long gid = (long)blockIdx.x * blockDim.x + threadIdx.x;
  if (gid >= (long)Mpad * 128) return;
  int row = (int)(gid >> 7);
  int c   = (int)(gid & 127);
  float v = 0.0f;
  if (row < M) {
    float cn = cnt[row];
    v = psum[(long)row * 128 + c] / (cn > 1.0f ? cn : 1.0f);
  }
  out[(long)row * ldo + c] = (bf16)v;
}

__global__ void feat_fill_kernel(const float* __restrict__ feat, bf16* __restrict__ A0,
                                 int M, int Mpad) {
  long gid = (long)blockIdx.x * blockDim.x + threadIdx.x;
  if (gid >= (long)Mpad * 32) return;
  int row = (int)(gid >> 5);
  int j   = (int)(gid & 31);
  float v = (row < M && j < FEAT) ? feat[(long)row * FEAT + j] : 0.0f;
  A0[(long)row * K_SAGE0 + 896 + j] = (bf16)v;
}

// ---------------- bf16 WMMA GEMM with async-LDS staged B ----------------
// C[m, coloff+n] = act( sum_k A[m,k]*Bt[n,k] + bias[n] )
// Block = 128 threads (4 waves). Block tile = 64x64, wave tile = 16x64.
// B tile (64 n-rows x 32 k) is staged once per block into LDS via
// GLOBAL_LOAD_ASYNC_TO_LDS_B128 (double-buffered, ASYNCcnt-paced).
// The last K-step is peeled so the steady-state loop body is branch-free:
// stage(next) -> A-load -> s_wait_asynccnt 2 -> barrier -> 4x WMMA -> barrier.
// LDS row stride 40 bf16 (80B) puts the 16-lane x 32B fragment reads on a
// conflict-free stride-4 bank lattice.
// ACT: 0=none 1=relu 2=tanh.  Requires K % 32 == 0 and K >= 64.
#define BS_STRIDE 40            // bf16 elements per staged row
template<int ACT, bool GATHER, bool STOREBF16>
__global__ __launch_bounds__(128)
void gemm_bf16_kernel(const bf16* __restrict__ A,
                      const float* __restrict__ Af32, const int* __restrict__ Aidx,
                      int idx_stride, int Mclamp, int lda,
                      const bf16* __restrict__ Bt, int ldb,
                      const float* __restrict__ bias,
                      float* __restrict__ Cf, bf16* __restrict__ Cb,
                      int ldc, int coloff, int M_store, int N, int K) {
  __shared__ __align__(16) bf16 Bs[2][64 * BS_STRIDE];

  const int tid     = threadIdx.x;
  const int lane    = tid & 31;
  const int wave    = tid >> 5;
  const int m0      = blockIdx.x * 64 + wave * 16;
  const int n0      = blockIdx.y * 64;
  const int halfsel = lane >> 4;     // 0: K 0..15 / rows 0..7 ; 1: K 16..31 / rows 8..15
  const int l15     = lane & 15;
  const int arow    = m0 + l15;
  const int kkbase  = halfsel << 4;

  // stage B tile [n0..n0+63] x [k0..k0+31] into Bs[p]; 2 x 16B per thread
  auto stage = [&](int p, int k0) {
#pragma unroll
    for (int i = 0; i < 2; ++i) {
      int chunk = tid * 2 + i;        // 0..255
      int row   = chunk >> 2;         // 0..63
      int part  = chunk & 3;          // 8 bf16 per part
      copy_b128_g2l(Bt + (long)(n0 + row) * ldb + k0 + part * 8,
                    &Bs[p][row * BS_STRIDE + part * 8]);
    }
  };

  // A fragment for K-chunk starting at k0 (contiguous k = 16h + s, same order as the B reads)
  auto load_a = [&](int k0) -> v16bf {
    const int kk = k0 + kkbase;
    v16bf a;
    if (GATHER) {
      int r = arow < Mclamp ? arow : (Mclamp - 1);
      const float* ap = Af32 + (long)Aidx[(long)r * idx_stride] * lda + kk;
#pragma unroll
      for (int q = 0; q < 4; ++q) {
        float4 f = ((const float4*)ap)[q];
        a[q * 4 + 0] = (bf16)f.x;
        a[q * 4 + 1] = (bf16)f.y;
        a[q * 4 + 2] = (bf16)f.z;
        a[q * 4 + 3] = (bf16)f.w;
      }
    } else {
      a = *(const v16bf*)(A + (long)arow * lda + kk);
    }
    return a;
  };

  v8f acc[4] = {v8f{}, v8f{}, v8f{}, v8f{}};

  auto mma_tile = [&](int p, const v16bf& a) {
#pragma unroll
    for (int j = 0; j < 4; ++j) {
      const bf16* bp = &Bs[p][(j * 16 + l15) * BS_STRIDE + kkbase];
      union { uint4 u[2]; v16bf v; } cvt;
      cvt.u[0] = *(const uint4*)bp;          // ds_load_b128
      cvt.u[1] = *(const uint4*)(bp + 8);    // ds_load_b128
      acc[j] = __builtin_amdgcn_wmma_f32_16x16x32_bf16(
          false, a, false, cvt.v, (short)0, acc[j], false, false);
    }
  };

  stage(0, 0);
  int p = 0;
  int k0 = 0;
  // steady state: branch-free body, last step peeled
  for (; k0 + 32 < K; k0 += 32) {
    stage(p ^ 1, k0 + 32);          // DMA next tile (ASYNCcnt += 2)
    v16bf a = load_a(k0);           // global A loads overlap the barrier
    wait_async_le2();               // current tile's copies complete
    __syncthreads();
    mma_tile(p, a);
    __syncthreads();                // protect Bs[p] before it is re-staged
    p ^= 1;
  }
  // peeled final step
  {
    v16bf a = load_a(k0);
    wait_async_le0();
    __syncthreads();
    mma_tile(p, a);
  }

#pragma unroll
  for (int j = 0; j < 4; ++j) {
    int col = n0 + j * 16 + l15;
    if (col >= N) continue;
    float bv = bias ? bias[col] : 0.0f;
#pragma unroll
    for (int r = 0; r < 8; ++r) {
      int row = m0 + halfsel * 8 + r;
      if (row >= M_store) continue;
      float v = acc[j][r] + bv;
      if (ACT == 1) v = v > 0.0f ? v : 0.0f;
      if (ACT == 2) v = tanhf(v);
      long off = (long)row * ldc + coloff + col;
      if (STOREBF16) Cb[off] = (bf16)v;
      else           Cf[off] = v;
    }
  }
}

// ---------------- launch ----------------
extern "C" void kernel_launch(void* const* d_in, const int* in_sizes, int n_in,
                              void* d_out, int out_size, void* d_ws, size_t ws_size,
                              hipStream_t stream) {
  const float* xa    = (const float*)d_in[0];
  const int*   ptok  = (const int*)  d_in[1];
  const float* pfeat = (const float*)d_in[2];
  const int*   ec    = (const int*)  d_in[3];   // [2, E]: row0 src, row1 dst
  const int*   wsrc  = (const int*)  d_in[4];
  const int*   wdst  = (const int*)  d_in[5];
  const float* emb   = (const float*)d_in[6];
  const float* Wp    = (const float*)d_in[7];
  const float* bp    = (const float*)d_in[8];
  const float* W0    = (const float*)d_in[9];
  const float* b0    = (const float*)d_in[10];
  // d_in[11], d_in[12] = gcn_W1/b1: dead code in the reference (a2 deleted)
  const float* Wl0   = (const float*)d_in[13];
  const float* Wr0   = (const float*)d_in[14];  // 780 x 128
  const float* sb0   = (const float*)d_in[15];
  const float* Wl1   = (const float*)d_in[16];
  const float* Wr1   = (const float*)d_in[17];
  const float* sb1   = (const float*)d_in[18];
  const float* linW  = (const float*)d_in[19];  // 128 x 64
  const float* linb  = (const float*)d_in[20];
  float* out = (float*)d_out;

  // ---- workspace carve (256B aligned) ----
  char* wp = (char*)d_ws;
  auto alloc = [&](size_t bytes) -> void* {
    void* r = (void*)wp;
    wp += (bytes + 255) & ~(size_t)255;
    return r;
  };
  float* dinv   = (float*)alloc(sizeof(float) * MA_PAD);            // deg -> dinv
  float* cnt    = (float*)alloc(sizeof(float) * MP_PAD);
  float* xw     = (float*)alloc(sizeof(float) * (size_t)MA_PAD * 128);
  float* agg    = (float*)alloc(sizeof(float) * (size_t)MA_PAD * 128);  // becomes a1
  float* psum   = (float*)alloc(sizeof(float) * (size_t)MP_PAD * 128);
  bf16*  xa_b   = (bf16*) alloc(sizeof(bf16)  * (size_t)MA_PAD * 128);
  bf16*  sageA0 = (bf16*) alloc(sizeof(bf16)  * (size_t)MP_PAD * K_SAGE0);
  bf16*  sageA1 = (bf16*) alloc(sizeof(bf16)  * (size_t)MP_PAD * K_SAGE1);
  bf16*  p2_b   = (bf16*) alloc(sizeof(bf16)  * (size_t)MP_PAD * 128);
  bf16*  Bt0    = (bf16*) alloc(sizeof(bf16)  * (size_t)128 * K_SAGE0);
  bf16*  Bt1    = (bf16*) alloc(sizeof(bf16)  * (size_t)128 * K_SAGE1);
  bf16*  W0t    = (bf16*) alloc(sizeof(bf16)  * (size_t)128 * 128);
  bf16*  Wpt    = (bf16*) alloc(sizeof(bf16)  * (size_t)D_BERT * D_BERT);
  bf16*  linWt  = (bf16*) alloc(sizeof(bf16)  * (size_t)OUT_DIM * 128);

  const int B = 256;
  const int* c_src = ec;
  const int* c_dst = ec + E_COLLAB;

  // ---- zero init of atomic targets ----
  zero_f32_kernel<<<ceil_div(MA_PAD, B), B, 0, stream>>>(dinv, MA_PAD);
  zero_f32_kernel<<<ceil_div(MP_PAD, B), B, 0, stream>>>(cnt, MP_PAD);
  zero_f32_kernel<<<ceil_div((long)MA_PAD * 128, B), B, 0, stream>>>(agg, (long)MA_PAD * 128);
  zero_f32_kernel<<<ceil_div((long)MP_PAD * 128, B), B, 0, stream>>>(psum, (long)MP_PAD * 128);

  // ---- degrees / counts, then dinv = rsqrt(deg+1) ----
  count_dst_kernel<<<ceil_div(E_COLLAB, B), B, 0, stream>>>(c_dst, dinv, E_COLLAB);
  count_dst_kernel<<<ceil_div(E_WRITES, B), B, 0, stream>>>(wdst, cnt, E_WRITES);
  rsqrt_kernel<<<ceil_div(N_AUTHOR, B), B, 0, stream>>>(dinv, N_AUTHOR);

  // ---- weight packs (transposed bf16) ----
  zero_bf16_kernel<<<ceil_div((long)128 * K_SAGE0, B), B, 0, stream>>>(Bt0, (long)128 * K_SAGE0);
  pack_bt_kernel<<<ceil_div((long)128 * 128, B), B, 0, stream>>>(W0,  W0t,  128,      0,   128, 128);
  pack_bt_kernel<<<ceil_div((long)D_BERT * D_BERT, B), B, 0, stream>>>(Wp, Wpt, D_BERT, 0, D_BERT, D_BERT);
  pack_bt_kernel<<<ceil_div((long)128 * 128, B), B, 0, stream>>>(Wl0, Bt0,  K_SAGE0,  0,   128, 128);
  pack_bt_kernel<<<ceil_div((long)780 * 128, B), B, 0, stream>>>(Wr0, Bt0,  K_SAGE0,  128, 780, 128);
  pack_bt_kernel<<<ceil_div((long)128 * 128, B), B, 0, stream>>>(Wl1, Bt1,  K_SAGE1,  0,   128, 128);
  pack_bt_kernel<<<ceil_div((long)128 * 128, B), B, 0, stream>>>(Wr1, Bt1,  K_SAGE1,  128, 128, 128);
  pack_bt_kernel<<<ceil_div((long)128 * OUT_DIM, B), B, 0, stream>>>(linW, linWt, 128, 0, 128, OUT_DIM);

  // ---- xa -> bf16 (pad rows zero) ----
  cvt_rows_kernel<<<ceil_div((long)MA_PAD * 128, B), B, 0, stream>>>(xa, xa_b, N_AUTHOR, MA_PAD, 128);

  // ---- GCN0: xw = xa @ W0 (WMMA) ----
  gemm_bf16_kernel<0, false, false><<<dim3(MA_PAD / 64, HID / 64), 128, 0, stream>>>(
      xa_b, nullptr, nullptr, 0, 0, 128, W0t, 128, nullptr,
      xw, nullptr, 128, 0, MA_PAD, HID, 128);
  scatter_gcn_kernel<<<ceil_div((long)E_COLLAB * 32, B), B, 0, stream>>>(
      c_src, c_dst, xw, dinv, agg, E_COLLAB);
  gcn_finalize_kernel<<<ceil_div((long)N_AUTHOR * 128, B), B, 0, stream>>>(
      agg, xw, dinv, b0, N_AUTHOR);   // agg now holds a1 (f32)

  // ---- SAGE0 aggregation: mean of xa over writes edges ----
  scatter_sum_kernel<<<ceil_div((long)E_WRITES * 32, B), B, 0, stream>>>(
      wsrc, wdst, xa, psum, E_WRITES);
  mean_cvt_kernel<<<ceil_div((long)MP_PAD * 128, B), B, 0, stream>>>(
      psum, cnt, sageA0, K_SAGE0, N_PAPER, MP_PAD);   // sageA0[:, 0:128]

  // ---- pooler = tanh(embed[tok0] @ Wp + bp) -> sageA0[:, 128:896] (fused gather, WMMA) ----
  gemm_bf16_kernel<2, true, true><<<dim3(MP_PAD / 64, D_BERT / 64), 128, 0, stream>>>(
      nullptr, emb, ptok, TOK, N_PAPER, D_BERT, Wpt, D_BERT, bp,
      nullptr, sageA0, K_SAGE0, 128, MP_PAD, D_BERT, D_BERT);

  // ---- paper_feat -> sageA0[:, 896:928) ----
  feat_fill_kernel<<<ceil_div((long)MP_PAD * 32, B), B, 0, stream>>>(pfeat, sageA0, N_PAPER, MP_PAD);

  // ---- SAGE0: p1 = relu([mean|xp] @ [Wl0;Wr0] + b0) -> sageA1[:, 128:256] ----
  gemm_bf16_kernel<1, false, true><<<dim3(MP_PAD / 64, HID / 64), 128, 0, stream>>>(
      sageA0, nullptr, nullptr, 0, 0, K_SAGE0, Bt0, K_SAGE0, sb0,
      nullptr, sageA1, K_SAGE1, 128, MP_PAD, HID, K_SAGE0);

  // ---- SAGE1 aggregation: mean of a1 over writes edges ----
  zero_f32_kernel<<<ceil_div((long)MP_PAD * 128, B), B, 0, stream>>>(psum, (long)MP_PAD * 128);
  scatter_sum_kernel<<<ceil_div((long)E_WRITES * 32, B), B, 0, stream>>>(
      wsrc, wdst, agg /*a1*/, psum, E_WRITES);
  mean_cvt_kernel<<<ceil_div((long)MP_PAD * 128, B), B, 0, stream>>>(
      psum, cnt, sageA1, K_SAGE1, N_PAPER, MP_PAD);   // sageA1[:, 0:128]

  // ---- SAGE1: p2 = relu([amean|p1] @ [Wl1;Wr1] + b1) ----
  gemm_bf16_kernel<1, false, true><<<dim3(MP_PAD / 64, HID / 64), 128, 0, stream>>>(
      sageA1, nullptr, nullptr, 0, 0, K_SAGE1, Bt1, K_SAGE1, sb1,
      nullptr, p2_b, 128, 0, MP_PAD, HID, K_SAGE1);

  // ---- out = p2 @ lin_W + lin_b (f32, only real rows) ----
  gemm_bf16_kernel<0, false, false><<<dim3(MP_PAD / 64, 1), 128, 0, stream>>>(
      p2_b, nullptr, nullptr, 0, 0, 128, linWt, 128, linb,
      out, nullptr, OUT_DIM, 0, N_PAPER, OUT_DIM, 128);
}
